// Transformer_9895604650465
// MI455X (gfx1250) — matrix operand order repaired and checked by reading, not compiled, not run on hardware
//
#include <hip/hip_runtime.h>
#include <hip/hip_bf16.h>
#include <stdint.h>

// Transformer encoder/decoder forward for MI455X (gfx1250, wave32, WMMA).
// GEMMs: v_wmma_f32_16x16x32_bf16; B tiles staged via global_load_async_to_lds
// (ASYNCcnt) double-buffered in LDS; A fragments register-pipelined.
// Wave tile 32x64 (2 A-frags x 4 N-tiles -> 8 WMMA per k-step, each LDS B
// fragment feeds two WMMAs). Block = 8 waves = 256x64 output tile.

#define DM   1024            // d_model
#define NH   16              // heads
#define DHD  64              // head dim
#define NL   6               // layers
#define FFD  4096            // ff hidden
#define BSZ  8               // batch
#define SEQ  512             // seq len
#define MROW (BSZ*SEQ)       // 4096 token rows
#define EPSL 1e-5f
#define BK   32              // k-step per stage
#define BPAD 8               // LDS row padding (bf16) -> bank-conflict-free

typedef __hip_bfloat16 bf16;
typedef __attribute__((ext_vector_type(16))) __bf16 v16bf;
typedef __attribute__((ext_vector_type(8)))  float  v8f;

struct Frag32B { uint4 lo, hi; };
struct BF8     { bf16 v[8]; };

// ---- WMMA A fragment (16x32 bf16) straight from global ---------------------
// lane r (0..15): row M=r, K={0..7},{16..23}; lane r+16: K={8..15},{24..31}.
__device__ __forceinline__ v16bf load_a_frag(const bf16* __restrict__ base, int ld,
                                             int row, int k0, int lane) {
  int half = lane >> 4, r = lane & 15;
  const uint4* p = (const uint4*)(base + (size_t)(row + r) * ld + k0 + half * 8);
  Frag32B f; f.lo = p[0]; f.hi = p[2];   // +16 elements = +2 uint4
  return __builtin_bit_cast(v16bf, f);
}

// ---- Generic batched bf16 GEMM: C = scale*(A @ B) + bias, optional ReLU ---
// A: [M,K] bf16 row-major, B: [K,N] bf16 row-major, C: f32 or bf16.
// Grid: (N/64, M/256, batch). Block: 256 threads = 8 waves; wave tile 32x64.
__global__ __launch_bounds__(256)
void k_gemm(const bf16* __restrict__ A, int lda, long long sA,
            const bf16* __restrict__ B, int ldb, long long sB,
            void* __restrict__ Cv, int ldc, long long sC,
            const float* __restrict__ bias,
            int K, float scale, int relu, int out_bf16) {
  __shared__ __align__(16) bf16 Bs[2][BK][64 + BPAD];

  const int tid   = threadIdx.x;
  const int lane  = tid & 31;
  const int wave  = tid >> 5;
  const int tileM = blockIdx.y * 256 + wave * 32;
  const int tileN = blockIdx.x * 64;
  A += (size_t)blockIdx.z * sA;
  B += (size_t)blockIdx.z * sB;

  // Cooperative fill mapping: 256 threads x 16B == 32x64 bf16 tile.
  const int frow = tid >> 3;              // 0..31 (k within stage)
  const int fcol = (tid & 7) * 8;         // 0..56 (n within tile)
  const bf16* gB = B + (size_t)frow * ldb + tileN + fcol;

  auto async_fill = [&](int buf, int k0) {
    uint32_t ldsa = (uint32_t)(uintptr_t)(&Bs[buf][frow][fcol]);
    uint64_t ga   = (uint64_t)(uintptr_t)(gB + (size_t)k0 * ldb);
    asm volatile("global_load_async_to_lds_b128 %0, %1, off"
                 :: "v"(ldsa), "v"(ga) : "memory");
  };
  auto lds_b_frag = [&](int buf, int t) {
    // lane r (0..15): B col N=r, K={0..7},{16..23}; lane r+16: K={8..15},{24..31}.
    int half = lane >> 4, r = lane & 15;
    v16bf b;
#pragma unroll
    for (int s = 0; s < 16; ++s) {
      int kk = (s < 8) ? (half * 8 + s) : (16 + half * 8 + (s - 8));
      b[s] = __builtin_bit_cast(__bf16, Bs[buf][kk][t * 16 + r]);
    }
    return b;
  };

  v8f acc[2][4] = {};

  // Prologue: stage 0 in flight, first A fragments in registers.
  async_fill(0, 0);
  v16bf a0 = load_a_frag(A, lda, tileM,      0, lane);
  v16bf a1 = load_a_frag(A, lda, tileM + 16, 0, lane);
  asm volatile("s_wait_asynccnt 0x0" ::: "memory");
  __syncthreads();

  for (int k0 = 0; k0 < K; k0 += BK) {
    const int buf = (k0 >> 5) & 1;
    const bool nxt = (k0 + BK) < K;
    v16bf a0n = a0, a1n = a1;
    if (nxt) {                       // prefetch next stage while we compute
      async_fill(buf ^ 1, k0 + BK);
      a0n = load_a_frag(A, lda, tileM,      k0 + BK, lane);
      a1n = load_a_frag(A, lda, tileM + 16, k0 + BK, lane);
    }
    // Each LDS B fragment feeds two WMMAs (rows tileM and tileM+16).
#pragma unroll
    for (int t = 0; t < 4; ++t) {
      v16bf bt = lds_b_frag(buf, t);
      acc[0][t] = __builtin_amdgcn_wmma_f32_16x16x32_bf16(
          false, a0, false, bt, (short)0, acc[0][t], false, false);
      acc[1][t] = __builtin_amdgcn_wmma_f32_16x16x32_bf16(
          false, a1, false, bt, (short)0, acc[1][t], false, false);
    }
    if (nxt) {
      asm volatile("s_wait_asynccnt 0x0" ::: "memory");
      __syncthreads();
    }
    a0 = a0n; a1 = a1n;
  }

  // C/D layout: VGPR j -> lanes 0-15: M=j, N=lane; lanes 16-31: M=j+8, N=lane-16.
  const int half = lane >> 4;
  const int cn   = lane & 15;
  if (out_bf16) {
    bf16* C = (bf16*)Cv + (size_t)blockIdx.z * sC;
#pragma unroll
    for (int mi = 0; mi < 2; ++mi) {
      int rbase = tileM + mi * 16 + half * 8;
#pragma unroll
      for (int t = 0; t < 4; ++t) {
        int col = tileN + t * 16 + cn;
        float bv = bias ? bias[col] : 0.0f;
#pragma unroll
        for (int j = 0; j < 8; ++j) {
          float v = acc[mi][t][j] * scale + bv;
          if (relu) v = fmaxf(v, 0.0f);
          C[(size_t)(rbase + j) * ldc + col] = __float2bfloat16(v);
        }
      }
    }
  } else {
    float* C = (float*)Cv + (size_t)blockIdx.z * sC;
#pragma unroll
    for (int mi = 0; mi < 2; ++mi) {
      int rbase = tileM + mi * 16 + half * 8;
#pragma unroll
      for (int t = 0; t < 4; ++t) {
        int col = tileN + t * 16 + cn;
        float bv = bias ? bias[col] : 0.0f;
#pragma unroll
        for (int j = 0; j < 8; ++j) {
          float v = acc[mi][t][j] * scale + bv;
          if (relu) v = fmaxf(v, 0.0f);
          C[(size_t)(rbase + j) * ldc + col] = v;
        }
      }
    }
  }
}

// ---- f32 -> bf16 cast, 8 elements/thread (b128 in, b128 out) ---------------
__global__ void k_cast(const float* __restrict__ in, bf16* __restrict__ out, size_t n8) {
  size_t i = (size_t)blockIdx.x * 256 + threadIdx.x;
  if (i >= n8) return;
  const float4* p = (const float4*)in + 2 * i;
  float4 x = p[0], y = p[1];
  BF8 o;
  o.v[0] = __float2bfloat16(x.x); o.v[1] = __float2bfloat16(x.y);
  o.v[2] = __float2bfloat16(x.z); o.v[3] = __float2bfloat16(x.w);
  o.v[4] = __float2bfloat16(y.x); o.v[5] = __float2bfloat16(y.y);
  o.v[6] = __float2bfloat16(y.z); o.v[7] = __float2bfloat16(y.w);
  ((uint4*)out)[i] = __builtin_bit_cast(uint4, o);
}

// ---- f32 copy + bf16 cast (activation init), 8 elements/thread -------------
__global__ void k_copy_cast(const float* __restrict__ in, float* __restrict__ outF,
                            bf16* __restrict__ outB, size_t n8) {
  size_t i = (size_t)blockIdx.x * 256 + threadIdx.x;
  if (i >= n8) return;
  const float4* p = (const float4*)in + 2 * i;
  float4 x = p[0], y = p[1];
  ((float4*)outF)[2 * i] = x;
  ((float4*)outF)[2 * i + 1] = y;
  BF8 o;
  o.v[0] = __float2bfloat16(x.x); o.v[1] = __float2bfloat16(x.y);
  o.v[2] = __float2bfloat16(x.z); o.v[3] = __float2bfloat16(x.w);
  o.v[4] = __float2bfloat16(y.x); o.v[5] = __float2bfloat16(y.y);
  o.v[6] = __float2bfloat16(y.z); o.v[7] = __float2bfloat16(y.w);
  ((uint4*)outB)[i] = __builtin_bit_cast(uint4, o);
}

// ---- qkv [B,S,3D] -> Q [B,H,S,64], Kt [B,H,64,S], V [B,H,S,64] (bf16) ------
__global__ void k_split_heads(const bf16* __restrict__ qkv, bf16* __restrict__ q,
                              bf16* __restrict__ kt, bf16* __restrict__ v) {
  size_t idx = (size_t)blockIdx.x * 256 + threadIdx.x;   // over B*H*S*64
  int c = (int)(idx & 63);
  size_t r = idx >> 6;
  int s = (int)(r % SEQ);
  size_t r2 = r / SEQ;
  int h = (int)(r2 % NH);
  int b = (int)(r2 / NH);
  size_t base = ((size_t)(b * SEQ + s)) * (3 * DM) + h * DHD + c;
  q[idx] = qkv[base];
  kt[(((size_t)(b * NH + h)) * DHD + c) * SEQ + s] = qkv[base + DM];
  v[idx] = qkv[base + 2 * DM];
}

// ---- in-place row softmax over bf16 scores, rows of length SEQ=512 ---------
__global__ __launch_bounds__(256)
void k_softmax(bf16* __restrict__ s) {
  __shared__ float sh[256];
  bf16* p = s + (size_t)blockIdx.x * SEQ;
  int t = threadIdx.x;
  float v0 = __bfloat162float(p[t]);
  float v1 = __bfloat162float(p[t + 256]);
  sh[t] = fmaxf(v0, v1);
  __syncthreads();
  for (int o = 128; o > 0; o >>= 1) { if (t < o) sh[t] = fmaxf(sh[t], sh[t + o]); __syncthreads(); }
  float m = sh[0];
  __syncthreads();
  v0 = __expf(v0 - m); v1 = __expf(v1 - m);
  sh[t] = v0 + v1;
  __syncthreads();
  for (int o = 128; o > 0; o >>= 1) { if (t < o) sh[t] += sh[t + o]; __syncthreads(); }
  float inv = 1.0f / sh[0];
  p[t] = __float2bfloat16(v0 * inv);
  p[t + 256] = __float2bfloat16(v1 * inv);
}

// ---- LayerNorm over D=1024 with fused residual add + optional head merge ---
__global__ __launch_bounds__(256)
void k_layernorm(const float* __restrict__ resid, const float* __restrict__ inp,
                 int headMerge, const float* __restrict__ g, const float* __restrict__ be,
                 float* __restrict__ outF, bf16* __restrict__ outB) {
  __shared__ float sh[256];
  int row = blockIdx.x, t = threadIdx.x;
  int b = row / SEQ, s = row % SEQ;
  float vals[4];
#pragma unroll
  for (int i = 0; i < 4; ++i) {
    int d = t + i * 256;
    size_t ii;
    if (headMerge) {
      int h = d >> 6, c = d & 63;
      ii = (((size_t)(b * NH + h)) * SEQ + s) * DHD + c;
    } else {
      ii = (size_t)row * DM + d;
    }
    float x = inp[ii];
    if (resid) x += resid[(size_t)row * DM + d];
    vals[i] = x;
  }
  float sum = vals[0] + vals[1] + vals[2] + vals[3];
  sh[t] = sum; __syncthreads();
  for (int o = 128; o > 0; o >>= 1) { if (t < o) sh[t] += sh[t + o]; __syncthreads(); }
  float mean = sh[0] * (1.0f / DM);
  __syncthreads();
  float ss = 0.0f;
#pragma unroll
  for (int i = 0; i < 4; ++i) { float dlt = vals[i] - mean; ss += dlt * dlt; }
  sh[t] = ss; __syncthreads();
  for (int o = 128; o > 0; o >>= 1) { if (t < o) sh[t] += sh[t + o]; __syncthreads(); }
  float rstd = rsqrtf(sh[0] * (1.0f / DM) + EPSL);
#pragma unroll
  for (int i = 0; i < 4; ++i) {
    int d = t + i * 256;
    float y = (vals[i] - mean) * rstd * g[d] + be[d];
    if (outF) outF[(size_t)row * DM + d] = y;
    if (outB) outB[(size_t)row * DM + d] = __float2bfloat16(y);
  }
}

// ---------------------------------------------------------------------------
extern "C" void kernel_launch(void* const* d_in, const int* in_sizes, int n_in,
                              void* d_out, int out_size, void* d_ws, size_t ws_size,
                              hipStream_t stream) {
  (void)in_sizes; (void)n_in; (void)out_size; (void)ws_size;
  const float* src          = (const float*)d_in[0];
  const float* tgt          = (const float*)d_in[1];
  const float* enc_attn_w   = (const float*)d_in[2];
  const float* enc_attn_b   = (const float*)d_in[3];
  const float* enc_ff_w1    = (const float*)d_in[4];
  const float* enc_ff_b1    = (const float*)d_in[5];
  const float* enc_ff_w2    = (const float*)d_in[6];
  const float* enc_ff_b2    = (const float*)d_in[7];
  const float* enc_ln_g     = (const float*)d_in[8];
  const float* enc_ln_b     = (const float*)d_in[9];
  const float* enc_out_ln_g = (const float*)d_in[10];
  const float* enc_out_ln_b = (const float*)d_in[11];
  const float* dec_attn_w   = (const float*)d_in[12];
  const float* dec_attn_b   = (const float*)d_in[13];
  const float* dec_ff_w1    = (const float*)d_in[14];
  const float* dec_ff_b1    = (const float*)d_in[15];
  const float* dec_ff_w2    = (const float*)d_in[16];
  const float* dec_ff_b2    = (const float*)d_in[17];
  const float* dec_ln_g     = (const float*)d_in[18];
  const float* dec_ln_b     = (const float*)d_in[19];
  const float* dec_out_ln_g = (const float*)d_in[20];
  const float* dec_out_ln_b = (const float*)d_in[21];

  float* out_mem = (float*)d_out;
  float* out_y   = out_mem + (size_t)MROW * DM;

  // Workspace carve-out (~160 MB), 256B aligned slices.
  char* p = (char*)d_ws;
  auto carve = [&](size_t bytes) -> void* {
    void* r = (void*)p; p += (bytes + 255) & ~((size_t)255); return r;
  };
  float* xf   = (float*)carve((size_t)MROW * DM * 4);          // activation f32
  bf16*  xb   = (bf16*) carve((size_t)MROW * DM * 2);          // activation bf16
  bf16*  wb   = (bf16*) carve((size_t)DM * FFD * 2);           // weight bf16 scratch
  bf16*  qkvb = (bf16*) carve((size_t)MROW * 3 * DM * 2);      // fused QKV bf16
  bf16*  qb   = (bf16*) carve((size_t)BSZ * NH * SEQ * DHD * 2);
  bf16*  ktb  = (bf16*) carve((size_t)BSZ * NH * DHD * SEQ * 2);
  bf16*  vb   = (bf16*) carve((size_t)BSZ * NH * SEQ * DHD * 2);
  bf16*  sc   = (bf16*) carve((size_t)BSZ * NH * SEQ * SEQ * 2); // scores (64MB)
  float* oh   = (float*)carve((size_t)BSZ * NH * SEQ * DHD * 4); // head outputs
  bf16*  hb   = sc;   // FF hidden (bf16, 32MB) reuses score buffer
  float* ffo  = oh;   // FF output (f32, 16MB) reuses head-output buffer

  auto gemm = [&](const bf16* A, int lda, long long sA,
                  const bf16* B, int ldb, long long sB,
                  void* C, int ldc, long long sC, const float* bias,
                  int M, int N, int K, float scale, int relu, int obf, int bat) {
    dim3 grid((unsigned)(N / 64), (unsigned)(M / 256), (unsigned)bat);
    k_gemm<<<grid, 256, 0, stream>>>(A, lda, sA, B, ldb, sB, C, ldc, sC,
                                     bias, K, scale, relu, obf);
  };
  auto cast = [&](const float* in, bf16* out, size_t n) {
    size_t n8 = n / 8;
    k_cast<<<dim3((unsigned)((n8 + 255) / 256)), 256, 0, stream>>>(in, out, n8);
  };
  auto lnorm = [&](const float* resid, const float* inp, int hm,
                   const float* g, const float* be, float* oF, bf16* oB) {
    k_layernorm<<<MROW, 256, 0, stream>>>(resid, inp, hm, g, be, oF, oB);
  };

  // Self-attention block: x = LN(x + Attn(x)); updates xf/xb in place.
  auto attn_block = [&](const float* w, const float* bias,
                        const float* g, const float* be) {
    cast(w, wb, (size_t)3 * DM * DM);
    for (int k = 0; k < 3; ++k)   // q/k/v projections share A, write fused qkv
      gemm(xb, DM, 0, wb + (size_t)k * DM * DM, DM, 0,
           qkvb + k * DM, 3 * DM, 0, bias + k * DM,
           MROW, DM, DM, 1.0f, 0, /*bf16 out*/1, 1);
    k_split_heads<<<(BSZ * NH * SEQ * DHD) / 256, 256, 0, stream>>>(qkvb, qb, ktb, vb);
    // scores = (Q @ K^T) / 8, bf16, batched over B*H
    gemm(qb, DHD, (long long)SEQ * DHD, ktb, SEQ, (long long)DHD * SEQ,
         sc, SEQ, (long long)SEQ * SEQ, nullptr,
         SEQ, SEQ, DHD, 0.125f, 0, 1, BSZ * NH);
    k_softmax<<<BSZ * NH * SEQ, 256, 0, stream>>>(sc);
    // O = P @ V, f32 head outputs
    gemm(sc, SEQ, (long long)SEQ * SEQ, vb, DHD, (long long)SEQ * DHD,
         oh, DHD, (long long)SEQ * DHD, nullptr,
         SEQ, DHD, SEQ, 1.0f, 0, 0, BSZ * NH);
    lnorm(xf, oh, /*headMerge*/1, g, be, xf, xb);
  };

  // Feed-forward block: x = LN(x + FF(x)); updates xf/xb in place.
  auto ff_block = [&](const float* w1, const float* b1,
                      const float* w2, const float* b2,
                      const float* g, const float* be) {
    cast(w1, wb, (size_t)DM * FFD);
    gemm(xb, DM, 0, wb, FFD, 0, hb, FFD, 0, b1,
         MROW, FFD, DM, 1.0f, /*relu*/1, /*bf16*/1, 1);
    cast(w2, wb, (size_t)FFD * DM);
    gemm(hb, FFD, 0, wb, DM, 0, ffo, DM, 0, b2,
         MROW, DM, FFD, 1.0f, 0, /*f32*/0, 1);
    lnorm(xf, ffo, 0, g, be, xf, xb);
  };

  // ---------------- Encoder ----------------
  k_copy_cast<<<(MROW * DM / 8) / 256, 256, 0, stream>>>(src, xf, xb, (size_t)MROW * DM / 8);
  for (int l = 0; l < NL; ++l) {
    attn_block(enc_attn_w + (size_t)l * 3 * DM * DM, enc_attn_b + (size_t)l * 3 * DM,
               enc_ln_g + (size_t)(l * 2 + 0) * DM, enc_ln_b + (size_t)(l * 2 + 0) * DM);
    ff_block(enc_ff_w1 + (size_t)l * DM * FFD, enc_ff_b1 + (size_t)l * FFD,
             enc_ff_w2 + (size_t)l * FFD * DM, enc_ff_b2 + (size_t)l * DM,
             enc_ln_g + (size_t)(l * 2 + 1) * DM, enc_ln_b + (size_t)(l * 2 + 1) * DM);
  }
  lnorm(nullptr, xf, 0, enc_out_ln_g, enc_out_ln_b, out_mem, nullptr);

  // ---------------- Decoder (faithful: never reads `memory`) ----------------
  k_copy_cast<<<(MROW * DM / 8) / 256, 256, 0, stream>>>(tgt, xf, xb, (size_t)MROW * DM / 8);
  for (int l = 0; l < NL; ++l) {
    for (int j = 0; j < 2; ++j)
      attn_block(dec_attn_w + (size_t)(l * 2 + j) * 3 * DM * DM,
                 dec_attn_b + (size_t)(l * 2 + j) * 3 * DM,
                 dec_ln_g + (size_t)(l * 3 + j) * DM,
                 dec_ln_b + (size_t)(l * 3 + j) * DM);
    ff_block(dec_ff_w1 + (size_t)l * DM * FFD, dec_ff_b1 + (size_t)l * FFD,
             dec_ff_w2 + (size_t)l * FFD * DM, dec_ff_b2 + (size_t)l * DM,
             dec_ln_g + (size_t)(l * 3 + 2) * DM, dec_ln_b + (size_t)(l * 3 + 2) * DM);
  }
  lnorm(nullptr, xf, 0, dec_out_ln_g, dec_out_ln_b, out_y, nullptr);
}
